// HybridRetention_52372831207858
// MI455X (gfx1250) — matrix operand order repaired and checked by reading, not compiled, not run on hardware
//
#include <hip/hip_runtime.h>
#include <hip/hip_bf16.h>
#include <math.h>

// Problem constants (match reference)
#define B_   2
#define S_   2048
#define DIM_ 1024
#define H_   16
#define HD_  64
#define BS_  (B_ * S_)   // 4096 rows
#define CH_  16          // scan chunks
#define CL_  (S_ / CH_)  // 128 steps per chunk

typedef __attribute__((ext_vector_type(16))) unsigned short v16us;
typedef __attribute__((ext_vector_type(16))) __bf16        v16bf;
typedef __attribute__((ext_vector_type(8)))  float         v8f;
typedef __attribute__((ext_vector_type(4)))  unsigned int  v4u;
typedef __attribute__((ext_vector_type(8)))  int           v8i;
typedef __attribute__((ext_vector_type(4)))  int           v4i;

#ifndef __has_builtin
#define __has_builtin(x) 0
#endif
#if __has_builtin(__builtin_amdgcn_tensor_load_to_lds) && \
    __has_builtin(__builtin_amdgcn_s_wait_tensorcnt)
#define USE_TDM 1
#else
#define USE_TDM 0
#endif

__device__ __forceinline__ unsigned short f2bf(float f) {
  unsigned int u = __float_as_uint(f);
  u += 0x7FFFu + ((u >> 16) & 1u);   // round-to-nearest-even
  return (unsigned short)(u >> 16);
}

__device__ __forceinline__ v16bf mk_bf16(v16us u) {
  return __builtin_bit_cast(v16bf, u);
}

// 16-bit A-matrix (16xK), contiguous k order used here: lanes 0-15 K=0..15, lanes 16-31 K=16..31
__device__ __forceinline__ int kmapA(int t, int laneHi) {
  return t + 16 * laneHi;
}
// 16-bit B-matrix (Kx16), contiguous k order used here: lanes 0-15 K=0..15, lanes 16-31 K=16..31
__device__ __forceinline__ int kmapB(int t, int laneHi) {
  return t + 16 * laneHi;
}

__device__ __forceinline__ v8f wmma_bf16(v16bf a, v16bf b, v8f c) {
  return __builtin_amdgcn_wmma_f32_16x16x32_bf16(
      false, a, false, b, (short)0, c, false, false);
}

// Force the compiler to assume LDS behind `p` was written (TDM writes are not
// visible to alias analysis, otherwise the ds_load fragments get undef-folded).
__device__ __forceinline__ void lds_written_fence(const void* p) {
  void* q = (void*)p;
  asm volatile("" : "+v"(q) : : "memory");
}

#if USE_TDM
// Issue a TDM 2-D tile load (bf16 elements) into LDS at byte offset lds_off.
// D# per CDNA5 ISA 8.3/8.4.
__device__ __forceinline__ void tdm_load_2d(unsigned lds_off, const void* gptr,
                                            unsigned tileK, unsigned tileRows,
                                            unsigned tensorK, unsigned tensorRows,
                                            unsigned rowStride) {
  unsigned long long ga = (unsigned long long)gptr;
  v4u g0;
  g0[0] = 1u;                                   // count=1
  g0[1] = lds_off;                              // lds_addr (bytes)
  g0[2] = (unsigned)(ga & 0xFFFFFFFFull);       // global_addr[31:0]
  g0[3] = (unsigned)((ga >> 32) & 0x01FFFFFFull) | (2u << 30);  // [56:32] | type=2
  v8i g1;
  g1[0] = (int)(1u << 16);                                      // data_size=1 (2B)
  g1[1] = (int)((tensorK & 0xFFFFu) << 16);                     // tdim0[15:0]
  g1[2] = (int)(((tensorK >> 16) & 0xFFFFu) | ((tensorRows & 0xFFFFu) << 16));
  g1[3] = (int)(((tensorRows >> 16) & 0xFFFFu) | ((tileK & 0xFFFFu) << 16));
  g1[4] = (int)(tileRows & 0xFFFFu);                            // tile_dim1
  g1[5] = (int)rowStride;                                       // tdim0_stride
  g1[6] = 0;
  g1[7] = 0;
  v4i z4 = {0, 0, 0, 0};
#if defined(__clang_major__) && (__clang_major__ >= 23)
  v8i z8 = {0, 0, 0, 0, 0, 0, 0, 0};
  __builtin_amdgcn_tensor_load_to_lds(g0, g1, z4, z4, z8, 0);
#else
  __builtin_amdgcn_tensor_load_to_lds(g0, g1, z4, z4, 0);
#endif
}
#endif  // USE_TDM

// ---------------------------------------------------------------------------
// fp32 -> bf16 (plain)
// ---------------------------------------------------------------------------
__global__ void hr_cvt_bf16(const float* __restrict__ in,
                            unsigned short* __restrict__ out, int n) {
  int i = blockIdx.x * blockDim.x + threadIdx.x;
  if (i < n) out[i] = f2bf(in[i]);
}

// fp32 [K,N] -> bf16 transposed [N,K]  (weights; one-time)
__global__ void hr_cvt_tr(const float* __restrict__ in,
                          unsigned short* __restrict__ out, int K, int N) {
  int i = blockIdx.x * blockDim.x + threadIdx.x;  // over N*K
  if (i < N * K) {
    int n = i / K, k = i % K;
    out[i] = f2bf(in[k * N + n]);
  }
}

// fp32 V[B,S,DIM] -> bf16 vT[B,H,HD,S]  (per-head transpose; one-time)
__global__ void hr_cvt_vT(const float* __restrict__ in,
                          unsigned short* __restrict__ out) {
  int i = blockIdx.x * blockDim.x + threadIdx.x;  // over B*H*HD*S
  int s   = i % S_;
  int d   = (i / S_) % HD_;
  int bh  = i / (S_ * HD_);
  int b   = bh / H_, h = bh % H_;
  out[i] = f2bf(in[(b * S_ + s) * DIM_ + h * HD_ + d]);
}

// ---------------------------------------------------------------------------
// Tiled BF16 GEMM: C[M,N] = A[M,K] * Bt[N,K]^T + bias, fp32 accumulate.
// Block tile 64x128, 8 waves (2x4), wave tile 32x32, BK=32.
// TDM path: double-buffered LDS, TDM prefetch of next K-tile overlapped with
// WMMA compute on current tile (s_wait_tensorcnt pipelining).
// mode 0: fp32 out; mode 1: sigmoid(fp32); mode 2: fp32 + bf16 out.
// ---------------------------------------------------------------------------
#define GBM 64
#define GBN 128
#define GBK 32
#define STAGE_E ((GBM + GBN) * GBK)   // elements per LDS stage

__global__ __launch_bounds__(256)
void hr_gemm_bf16(const unsigned short* __restrict__ A,
                  const unsigned short* __restrict__ Bt,
                  const float* __restrict__ bias,
                  float* __restrict__ Cf,
                  unsigned short* __restrict__ Cbf,
                  int M, int N, int K, int mode) {
  __shared__ unsigned short smem[2 * STAGE_E];

  const int tid    = threadIdx.x;
  const int lane   = tid & 31;
  const int wave   = tid >> 5;
  const int laneHi = lane >> 4;
  const int lcol   = lane & 15;
  const int wm     = (wave >> 2) * 32;  // 0 / 32
  const int wn     = (wave & 3) * 32;   // 0..96
  const int m0     = blockIdx.y * GBM;
  const int n0     = blockIdx.x * GBN;

  v8f acc[2][2];
#pragma unroll
  for (int i = 0; i < 2; ++i)
#pragma unroll
    for (int j = 0; j < 2; ++j)
#pragma unroll
      for (int e = 0; e < 8; ++e) acc[i][j][e] = 0.0f;

#if USE_TDM
  if (wave == 0) {  // preload stage 0
    tdm_load_2d(0u, A + (size_t)m0 * K, GBK, GBM, K, M, K);
    tdm_load_2d((unsigned)(GBM * GBK * 2), Bt + (size_t)n0 * K, GBK, GBN, K, N, K);
  }
  int stage = 0;
#else
  const unsigned int* Ag   = (const unsigned int*)A;
  const unsigned int* Bg   = (const unsigned int*)Bt;
  unsigned int*       As32 = (unsigned int*)smem;
  unsigned int*       Bs32 = (unsigned int*)(smem + GBM * GBK);
  const int Kd2 = K >> 1;
#endif

  for (int k0 = 0; k0 < K; k0 += GBK) {
#if USE_TDM
    if (wave == 0) __builtin_amdgcn_s_wait_tensorcnt(0);  // current stage ready
    lds_written_fence(smem);
    __syncthreads();
    if (wave == 0 && (k0 + GBK) < K) {  // prefetch next tile into other stage
      unsigned nb = (unsigned)((stage ^ 1) * STAGE_E * 2);
      tdm_load_2d(nb, A + (size_t)m0 * K + (k0 + GBK), GBK, GBM, K, M, K);
      tdm_load_2d(nb + (unsigned)(GBM * GBK * 2),
                  Bt + (size_t)n0 * K + (k0 + GBK), GBK, GBN, K, N, K);
    }
    const unsigned short* As = smem + stage * STAGE_E;   // [GBM][GBK]
    const unsigned short* Bs = As + GBM * GBK;           // [GBN][GBK]
#else
#pragma unroll
    for (int i = 0; i < 4; ++i) {
      int idx = i * 256 + tid;
      int r = idx >> 4, cp = idx & 15;
      As32[idx] = Ag[(m0 + r) * Kd2 + (k0 >> 1) + cp];
    }
#pragma unroll
    for (int i = 0; i < 8; ++i) {
      int idx = i * 256 + tid;
      int r = idx >> 4, cp = idx & 15;
      Bs32[idx] = Bg[(n0 + r) * Kd2 + (k0 >> 1) + cp];
    }
    const unsigned short* As = smem;
    const unsigned short* Bs = smem + GBM * GBK;
    __syncthreads();
#endif

    v16bf afr[2], bfr[2];
#pragma unroll
    for (int i = 0; i < 2; ++i) {
      int row = wm + i * 16 + lcol;
      v16us u;
#pragma unroll
      for (int t = 0; t < 16; ++t) u[t] = As[row * GBK + kmapA(t, laneHi)];
      afr[i] = mk_bf16(u);
    }
#pragma unroll
    for (int j = 0; j < 2; ++j) {
      int col = wn + j * 16 + lcol;
      v16us u;  // contiguous 16 ushorts: 2x ds_load_b128
#pragma unroll
      for (int t = 0; t < 16; ++t) u[t] = Bs[col * GBK + kmapB(t, laneHi)];
      bfr[j] = mk_bf16(u);
    }
#pragma unroll
    for (int i = 0; i < 2; ++i)
#pragma unroll
      for (int j = 0; j < 2; ++j)
        acc[i][j] = wmma_bf16(afr[i], bfr[j], acc[i][j]);
    __syncthreads();
#if USE_TDM
    stage ^= 1;
#endif
  }

  // Epilogue: C layout row = v + 8*laneHi, col = lane&15
#pragma unroll
  for (int i = 0; i < 2; ++i)
#pragma unroll
    for (int j = 0; j < 2; ++j) {
      int col = n0 + wn + j * 16 + lcol;
      float bv = bias[col];
#pragma unroll
      for (int v = 0; v < 8; ++v) {
        int row = m0 + wm + i * 16 + v + 8 * laneHi;
        float o = acc[i][j][v] + bv;
        if (mode == 1) o = 1.0f / (1.0f + __expf(-o));
        Cf[row * N + col] = o;
        if (mode == 2) Cbf[row * N + col] = f2bf(o);
      }
    }
}

// ---------------------------------------------------------------------------
// Chunked gated scan h_t = a_t*h_{t-1} + u_t  (3 phases, 16 chunks of 128)
// ---------------------------------------------------------------------------
__global__ void hr_scan_part1(const float* __restrict__ a,
                              const float* __restrict__ u,
                              float* __restrict__ carryP,
                              float* __restrict__ carryH) {
  int idx = blockIdx.x * blockDim.x + threadIdx.x;  // B*CH*DIM
  int d = idx % DIM_;
  int c = (idx / DIM_) % CH_;
  int b = idx / (DIM_ * CH_);
  int base = (b * S_ + c * CL_) * DIM_ + d;
  float P = 1.0f, h = 0.0f;
  for (int s = 0; s < CL_; ++s) {
    float av = a[base + s * DIM_];
    h = av * h + u[base + s * DIM_];
    P *= av;
  }
  carryP[idx] = P;
  carryH[idx] = h;
}

__global__ void hr_scan_part2(const float* __restrict__ carryP,
                              const float* __restrict__ carryH,
                              float* __restrict__ carryS) {
  int idx = blockIdx.x * blockDim.x + threadIdx.x;  // B*DIM
  int d = idx % DIM_, b = idx / DIM_;
  float st = 0.0f;
  for (int c = 0; c < CH_; ++c) {
    int o = (b * CH_ + c) * DIM_ + d;
    carryS[o] = st;
    st = carryP[o] * st + carryH[o];
  }
}

__global__ void hr_scan_part3(const float* __restrict__ a,
                              const float* __restrict__ u,
                              const float* __restrict__ carryS,
                              unsigned short* __restrict__ hbf) {
  int idx = blockIdx.x * blockDim.x + threadIdx.x;  // B*CH*DIM
  int d = idx % DIM_;
  int c = (idx / DIM_) % CH_;
  int b = idx / (DIM_ * CH_);
  int base = (b * S_ + c * CL_) * DIM_ + d;
  float h = carryS[(b * CH_ + c) * DIM_ + d];
  for (int s = 0; s < CL_; ++s) {
    int o = base + s * DIM_;
    h = a[o] * h + u[o];
    hbf[o] = f2bf(h);
  }
}

// ---------------------------------------------------------------------------
// Retention: per (b,h) & 16-query tile, 32-key steps, all matmuls via WMMA.
// Q[B,S,DIM] bf16, K[B,S,DIM] bf16, V transposed vT[B,H,HD,S] bf16 so all
// fragment loads are contiguous b128s.  Masked scores staged through LDS.
// ---------------------------------------------------------------------------
__global__ __launch_bounds__(32)
void hr_retention(const unsigned short* __restrict__ Q,
                  const unsigned short* __restrict__ Kb,
                  const unsigned short* __restrict__ Vt,
                  const float* __restrict__ gammas,
                  float* __restrict__ Out) {
  __shared__ unsigned short Sld[16 * 32];

  const int lane   = threadIdx.x;
  const int laneHi = lane >> 4;
  const int lcol   = lane & 15;
  const int n0     = blockIdx.x * 16;
  const int bh     = blockIdx.y;
  const int b      = bh / H_, h = bh % H_;
  const float lg   = log2f(gammas[h]);

  const unsigned short* qb = Q  + (b * S_ + n0) * DIM_ + h * HD_;
  const unsigned short* kb = Kb + b * S_ * DIM_ + h * HD_;
  const unsigned short* vb = Vt + (size_t)(b * H_ + h) * HD_ * S_;  // [HD][S]

  v16bf qf[2];
#pragma unroll
  for (int ks = 0; ks < 2; ++ks) {
    v16us u;
#pragma unroll
    for (int t = 0; t < 16; ++t)
      u[t] = qb[lcol * DIM_ + ks * 32 + kmapA(t, laneHi)];
    qf[ks] = mk_bf16(u);
  }

  v8f o[4];
#pragma unroll
  for (int j = 0; j < 4; ++j)
#pragma unroll
    for (int e = 0; e < 8; ++e) o[j][e] = 0.0f;

  const int mEnd = n0 + 16;
  for (int mblk = 0; mblk < mEnd; mblk += 32) {
#pragma unroll
    for (int tIdx = 0; tIdx < 2; ++tIdx) {
      v8f sc;
#pragma unroll
      for (int e = 0; e < 8; ++e) sc[e] = 0.0f;
#pragma unroll
      for (int ks = 0; ks < 2; ++ks) {
        v16us u;  // K^T fragment: contiguous along hd
#pragma unroll
        for (int t = 0; t < 16; ++t)
          u[t] = kb[(mblk + tIdx * 16 + lcol) * DIM_ + ks * 32 + kmapB(t, laneHi)];
        sc = wmma_bf16(qf[ks], mk_bf16(u), sc);
      }
#pragma unroll
      for (int v = 0; v < 8; ++v) {
        int row  = v + 8 * laneHi;
        int n    = n0 + row;
        int m    = mblk + tIdx * 16 + lcol;
        int diff = n - m;
        float dec = (diff >= 0) ? exp2f(lg * (float)diff) : 0.0f;
        Sld[row * 32 + tIdx * 16 + lcol] = f2bf(sc[v] * 0.125f * dec);
      }
    }
    __syncthreads();

    v16us su;
#pragma unroll
    for (int t = 0; t < 16; ++t) su[t] = Sld[lcol * 32 + kmapA(t, laneHi)];
    v16bf sf = mk_bf16(su);

#pragma unroll
    for (int j = 0; j < 4; ++j) {
      v16us u;  // V fragment from vT: contiguous along s
#pragma unroll
      for (int t = 0; t < 16; ++t)
        u[t] = vb[(j * 16 + lcol) * S_ + mblk + kmapB(t, laneHi)];
      o[j] = wmma_bf16(sf, mk_bf16(u), o[j]);
    }
    __syncthreads();
  }

#pragma unroll
  for (int j = 0; j < 4; ++j)
#pragma unroll
    for (int v = 0; v < 8; ++v) {
      int row = n0 + v + 8 * laneHi;
      Out[(b * S_ + row) * DIM_ + h * HD_ + j * 16 + lcol] = o[j][v];
    }
}

// ---------------------------------------------------------------------------
extern "C" void kernel_launch(void* const* d_in, const int* in_sizes, int n_in,
                              void* d_out, int out_size, void* d_ws, size_t ws_size,
                              hipStream_t stream) {
  (void)in_sizes; (void)n_in; (void)out_size; (void)ws_size;

  const float* q      = (const float*)d_in[0];
  const float* k      = (const float*)d_in[1];
  const float* v      = (const float*)d_in[2];
  const float* W_in   = (const float*)d_in[3];
  const float* b_in   = (const float*)d_in[4];
  const float* W_gate = (const float*)d_in[5];
  const float* b_gate = (const float*)d_in[6];
  const float* W_out  = (const float*)d_in[7];
  const float* b_out  = (const float*)d_in[8];
  const float* gammas = (const float*)d_in[9];
  float* out = (float*)d_out;

  char* ws = (char*)d_ws;
  size_t off = 0;
  auto alloc = [&](size_t bytes) -> void* {
    void* p = ws + off;
    off += (bytes + 255) & ~(size_t)255;
    return p;
  };
  const size_t actE = (size_t)BS_ * DIM_;   // 4M
  const size_t wE   = (size_t)DIM_ * DIM_;  // 1M

  unsigned short* x_bf     = (unsigned short*)alloc(actE * 2);
  unsigned short* k_bf     = (unsigned short*)alloc(actE * 2);
  unsigned short* vT_bf    = (unsigned short*)alloc(actE * 2);
  unsigned short* winT_bf  = (unsigned short*)alloc(wE * 2);
  unsigned short* wgateT_bf= (unsigned short*)alloc(wE * 2);
  unsigned short* woutT_bf = (unsigned short*)alloc(wE * 2);
  float*          u_f      = (float*)alloc(actE * 4);
  float*          a_f      = (float*)alloc(actE * 4);
  unsigned short* h_bf     = (unsigned short*)alloc(actE * 2);
  unsigned short* qproj_bf = (unsigned short*)alloc(actE * 2);
  float*          carryP   = (float*)alloc((size_t)B_ * CH_ * DIM_ * 4);
  float*          carryH   = (float*)alloc((size_t)B_ * CH_ * DIM_ * 4);
  float*          carryS   = (float*)alloc((size_t)B_ * CH_ * DIM_ * 4);

  const int TB = 256;
  hr_cvt_bf16<<<(int)(actE / TB), TB, 0, stream>>>(q, x_bf, (int)actE);
  hr_cvt_bf16<<<(int)(actE / TB), TB, 0, stream>>>(k, k_bf, (int)actE);
  hr_cvt_vT  <<<(int)(actE / TB), TB, 0, stream>>>(v, vT_bf);
  hr_cvt_tr  <<<(int)(wE / TB), TB, 0, stream>>>(W_in,   winT_bf,   DIM_, DIM_);
  hr_cvt_tr  <<<(int)(wE / TB), TB, 0, stream>>>(W_gate, wgateT_bf, DIM_, DIM_);
  hr_cvt_tr  <<<(int)(wE / TB), TB, 0, stream>>>(W_out,  woutT_bf,  DIM_, DIM_);

  dim3 gGrid(DIM_ / GBN, BS_ / GBM);  // (8, 64)
  hr_gemm_bf16<<<gGrid, 256, 0, stream>>>(x_bf, winT_bf, b_in, u_f, nullptr,
                                          BS_, DIM_, DIM_, 0);
  hr_gemm_bf16<<<gGrid, 256, 0, stream>>>(x_bf, wgateT_bf, b_gate, a_f, nullptr,
                                          BS_, DIM_, DIM_, 1);

  hr_scan_part1<<<(B_ * CH_ * DIM_) / TB, TB, 0, stream>>>(a_f, u_f, carryP, carryH);
  hr_scan_part2<<<(B_ * DIM_) / TB, TB, 0, stream>>>(carryP, carryH, carryS);
  hr_scan_part3<<<(B_ * CH_ * DIM_) / TB, TB, 0, stream>>>(a_f, u_f, carryS, h_bf);

  hr_gemm_bf16<<<gGrid, 256, 0, stream>>>(h_bf, woutT_bf, b_out, u_f, qproj_bf,
                                          BS_, DIM_, DIM_, 2);

  dim3 rGrid(S_ / 16, B_ * H_);  // (128, 32)
  hr_retention<<<rGrid, 32, 0, stream>>>(qproj_bf, k_bf, vT_bf, gammas, out);
}
